// HindiCausalLMAttention_61418032333135
// MI455X (gfx1250) — matrix operand order repaired and checked by reading, not compiled, not run on hardware
//
#include <hip/hip_runtime.h>

// ---------------------------------------------------------------------------
// Problem constants (match reference)
// ---------------------------------------------------------------------------
static constexpr int S    = 2048;
static constexpr int HID  = 4096;
static constexpr int NH   = 32;
static constexpr int NKV  = 8;
static constexpr int HD   = 128;
static constexpr int NQ   = NH * HD;    // 4096
static constexpr int NKVD = NKV * HD;   // 1024

typedef __attribute__((ext_vector_type(16))) __bf16 bf16x16;
typedef __attribute__((ext_vector_type(8)))  float  f32x8;

union FragU {
  bf16x16 v;
  uint4   q[2];
};

// ---------------------------------------------------------------------------
// bf16 helpers (round-to-nearest-even)
// ---------------------------------------------------------------------------
__device__ inline unsigned short f2bf(float f) {
  unsigned u = __float_as_uint(f);
  unsigned r = u + 0x7fffu + ((u >> 16) & 1u);
  return (unsigned short)(r >> 16);
}
__device__ inline float bf2f(unsigned short u) {
  return __uint_as_float(((unsigned)u) << 16);
}

// ---------------------------------------------------------------------------
// CDNA5 async memory->LDS copy (ASYNCcnt-tracked, bypasses VGPRs).
// LDS destination offset is taken from the vdst VGPR; generic-pointer low 32 bits == LDS offset
// (aperture tag lives entirely in addr[63:32] on gfx1250).
// ---------------------------------------------------------------------------
__device__ inline void async_copy16(void* lds_dst, const void* gsrc) {
  unsigned lds_off = (unsigned)(size_t)lds_dst;
  unsigned long long gaddr = (unsigned long long)(size_t)gsrc;
  asm volatile("global_load_async_to_lds_b128 %0, %1, off"
               :: "v"(lds_off), "v"(gaddr)
               : "memory");
}
__device__ inline void wait_async0() {
  asm volatile("s_wait_asynccnt 0x0" ::: "memory");
}

// ---------------------------------------------------------------------------
// WMMA fragment loads (CDNA5 16-bit layouts, wave32)
// A (16x32, MxK): lane<16 -> row=lane, elems K0..7 and K16..23
//                 lane>=16 -> row=lane-16, elems K8..15 and K24..31
// B (32x16, KxN): lane<16 -> col=lane, elems K0..7 and K16..23
//                 lane>=16 -> col=lane-16, elems K8..15 and K24..31
// ---------------------------------------------------------------------------
__device__ inline bf16x16 load_frag_a(const unsigned short* p, int lda) {
  const int lane = threadIdx.x & 31;
  const unsigned short* q = p + (lane & 15) * lda + ((lane >> 4) << 3);
  FragU f;
  f.q[0] = *(const uint4*)(q);
  f.q[1] = *(const uint4*)(q + 16);
  return f.v;
}
__device__ inline bf16x16 load_frag_b(const unsigned short* p, int ldb) {
  const int lane = threadIdx.x & 31;
  const unsigned short* q = p + (lane & 15) * ldb + ((lane >> 4) << 3);
  FragU f;
  f.q[0] = *(const uint4*)(q);
  f.q[1] = *(const uint4*)(q + 16);
  return f.v;
}
__device__ inline f32x8 wmma_bf16(bf16x16 a, bf16x16 b, f32x8 c) {
  return __builtin_amdgcn_wmma_f32_16x16x32_bf16(false, a, false, b,
                                                 (short)0, c, false, false);
}

// ---------------------------------------------------------------------------
// 16-lane-half row reductions via DPP16 ROW_XMASK (0x160|mask): pure-VALU
// xor-butterfly inside each 16-lane row half -- no LDS (ds_bpermute) traffic,
// co-issues against the WMMA pipe. C/D layout keeps one output row per half.
// ---------------------------------------------------------------------------
template <int CTRL>
__device__ inline float dpp_xor(float v) {
  int r = __builtin_amdgcn_update_dpp(0, __float_as_int(v), CTRL, 0xf, 0xf, true);
  return __int_as_float(r);
}
__device__ inline float rmax16(float v) {
  v = fmaxf(v, dpp_xor<0x161>(v));   // row_xmask:1
  v = fmaxf(v, dpp_xor<0x162>(v));   // row_xmask:2
  v = fmaxf(v, dpp_xor<0x164>(v));   // row_xmask:4
  v = fmaxf(v, dpp_xor<0x168>(v));   // row_xmask:8
  return v;
}
__device__ inline float rsum16(float v) {
  v += dpp_xor<0x161>(v);
  v += dpp_xor<0x162>(v);
  v += dpp_xor<0x164>(v);
  v += dpp_xor<0x168>(v);
  return v;
}

// ---------------------------------------------------------------------------
// fp32 -> bf16 conversion
// ---------------------------------------------------------------------------
__global__ void cvt_f32_bf16(const float* __restrict__ in,
                             unsigned short* __restrict__ out, int n) {
  int i = blockIdx.x * blockDim.x + threadIdx.x;
  if (i < n) out[i] = f2bf(in[i]);
}

// ---------------------------------------------------------------------------
// Generic GEMM: C[M,N] = A[M,K](bf16) * W[N,K](bf16)^T
// block = 256 threads (8 waves), block tile 64x128, wave tile 32x32.
// Tiles staged with async global->LDS copies, double buffered: iteration ik
// computes WMMAs from buffer `cur` while async loads fill buffer `cur^1`.
// ---------------------------------------------------------------------------
template <int OUT_BF16>
__global__ __launch_bounds__(256) void gemm_xwt(
    const unsigned short* __restrict__ A, const unsigned short* __restrict__ W,
    void* __restrict__ Cout, int M, int N, int K) {
  __shared__ __align__(16) unsigned short As[2][64 * 32];
  __shared__ __align__(16) unsigned short Ws[2][128 * 32];

  const int t    = threadIdx.x;
  const int lane = t & 31;
  const int w    = t >> 5;
  const int m0   = blockIdx.y * 64;
  const int n0   = blockIdx.x * 128;
  const int wm   = (w >> 2) * 32;   // 0 / 32
  const int wn   = (w & 3) * 32;    // 0..96

  f32x8 c00 = {}, c01 = {}, c10 = {}, c11 = {};

  const int a_row = t >> 2, a_k = (t & 3) << 3;   // 64 rows x 4 segs of 8
  const int w_row = t >> 1, w_k = (t & 1) << 4;   // 128 rows x 2 segs of 16

  auto stage = [&](int kk, int buf) {
    async_copy16(&As[buf][a_row * 32 + a_k],     &A[(m0 + a_row) * K + kk + a_k]);
    async_copy16(&Ws[buf][w_row * 32 + w_k],     &W[(n0 + w_row) * K + kk + w_k]);
    async_copy16(&Ws[buf][w_row * 32 + w_k + 8], &W[(n0 + w_row) * K + kk + w_k + 8]);
  };

  const int nk = K / 32;
  stage(0, 0);
  wait_async0();
  __syncthreads();

  int cur = 0;
  for (int ik = 0; ik < nk; ++ik) {
    if (ik + 1 < nk) stage((ik + 1) * 32, cur ^ 1);

    bf16x16 a0 = load_frag_a(&As[cur][(wm + 0)  * 32], 32);
    bf16x16 a1 = load_frag_a(&As[cur][(wm + 16) * 32], 32);
    bf16x16 b0 = load_frag_b(&Ws[cur][(wn + 0)  * 32], 32);
    bf16x16 b1 = load_frag_b(&Ws[cur][(wn + 16) * 32], 32);
    c00 = wmma_bf16(a0, b0, c00);
    c01 = wmma_bf16(a0, b1, c01);
    c10 = wmma_bf16(a1, b0, c10);
    c11 = wmma_bf16(a1, b1, c11);

    wait_async0();
    __syncthreads();
    cur ^= 1;
  }

  const int cn = lane & 15;
  const int r0 = (lane >> 4) << 3;
#pragma unroll
  for (int tm = 0; tm < 2; ++tm) {
#pragma unroll
    for (int tn = 0; tn < 2; ++tn) {
      f32x8 c = (tm == 0) ? (tn == 0 ? c00 : c01) : (tn == 0 ? c10 : c11);
#pragma unroll
      for (int i = 0; i < 8; ++i) {
        int row = m0 + wm + tm * 16 + r0 + i;
        int col = n0 + wn + tn * 16 + cn;
        if (OUT_BF16) ((unsigned short*)Cout)[row * N + col] = f2bf(c[i]);
        else          ((float*)Cout)[row * N + col] = c[i];
      }
    }
  }
}

// ---------------------------------------------------------------------------
// RoPE + per-head layout: lin[S][nheads*HD](bf16) -> out[h][S][HD](bf16)
// ---------------------------------------------------------------------------
__global__ void rope_layout(const unsigned short* __restrict__ lin,
                            const float* __restrict__ cosb,
                            const float* __restrict__ sinb,
                            unsigned short* __restrict__ out, int nheads) {
  int idx = blockIdx.x * blockDim.x + threadIdx.x;
  int total = S * nheads * HD;
  if (idx >= total) return;
  int d = idx & (HD - 1);
  int h = (idx / HD) % nheads;
  int s = idx / (HD * nheads);
  int base = s * nheads * HD + h * HD;
  float v = bf2f(lin[base + d]);
  float partner = (d < HD / 2) ? -bf2f(lin[base + d + HD / 2])
                               :  bf2f(lin[base + d - HD / 2]);
  float r = v * cosb[s * HD + d] + partner * sinb[s * HD + d];
  out[((size_t)h * S + s) * HD + d] = f2bf(r);
}

// V layout: lin[S][NKV*HD] -> Vt[kv][HD][S] (transposed for contiguous B-frags)
__global__ void v_layout(const unsigned short* __restrict__ lin,
                         unsigned short* __restrict__ out) {
  int idx = blockIdx.x * blockDim.x + threadIdx.x;
  int total = S * NKV * HD;
  if (idx >= total) return;
  int d = idx & (HD - 1);
  int h = (idx / HD) % NKV;
  int s = idx / (HD * NKV);
  out[((size_t)h * HD + d) * S + s] = lin[idx];
}

// ---------------------------------------------------------------------------
// Flash attention: each wave owns 16 query rows of one head; streams causal
// key chunks of 32; online softmax in registers; P re-layout via per-wave LDS.
// grid = (NH, S/64), block = 128 (4 waves)
// ---------------------------------------------------------------------------
__global__ __launch_bounds__(128) void flash_attn(
    const unsigned short* __restrict__ Qh, const unsigned short* __restrict__ Kh,
    const unsigned short* __restrict__ Vt, unsigned short* __restrict__ Ob) {
  __shared__ __align__(16) unsigned short Pls[4][16 * 32];

  const int t     = threadIdx.x;
  const int w     = t >> 5;
  const int lane  = t & 31;
  const int h     = blockIdx.x;
  const int kvh   = h >> 2;                       // NH/NKV = 4
  const int qbase = (blockIdx.y * 4 + w) * 16;

  const unsigned short* Qbase = Qh + ((size_t)h * S + qbase) * HD;
  const unsigned short* Kbase = Kh + (size_t)kvh * S * HD;
  const unsigned short* Vbase = Vt + (size_t)kvh * HD * S;
  unsigned short* P = &Pls[w][0];

  bf16x16 qf[4];
#pragma unroll
  for (int d = 0; d < 4; ++d) qf[d] = load_frag_a(Qbase + d * 32, HD);

  f32x8 acc[8];
  f32x8 zero = {};
#pragma unroll
  for (int i = 0; i < 8; ++i) acc[i] = zero;

  float mrow[8], lrow[8];
#pragma unroll
  for (int i = 0; i < 8; ++i) { mrow[i] = -1e30f; lrow[i] = 0.0f; }

  const int   cn    = lane & 15;
  const int   r0    = (lane >> 4) << 3;
  const float scale = 0.08838834764831845f;       // 1/sqrt(128)

  for (int kb = 0; kb <= qbase; kb += 32) {
    // ---- scores 16x32 = Q(16xHD) . K^T -------------------------------
    f32x8 s0 = zero, s1 = zero;
#pragma unroll
    for (int d = 0; d < 4; ++d) {
      bf16x16 b0 = load_frag_b(Kbase + (size_t)kb * HD + d * 32, HD);
      bf16x16 b1 = load_frag_b(Kbase + (size_t)(kb + 16) * HD + d * 32, HD);
      s0 = wmma_bf16(qf[d], b0, s0);
      s1 = wmma_bf16(qf[d], b1, s1);
    }
    // ---- scale + causal mask (wave-uniform guard) --------------------
    const bool need_mask = (kb + 31 > qbase);
#pragma unroll
    for (int i = 0; i < 8; ++i) {
      s0[i] *= scale;
      s1[i] *= scale;
    }
    if (need_mask) {
#pragma unroll
      for (int i = 0; i < 8; ++i) {
        int qi = qbase + r0 + i;
        if (kb + cn > qi)      s0[i] = -1e9f;
        if (kb + 16 + cn > qi) s1[i] = -1e9f;
      }
    }
    // ---- online softmax ----------------------------------------------
#pragma unroll
    for (int i = 0; i < 8; ++i) {
      float mnew = fmaxf(mrow[i], rmax16(fmaxf(s0[i], s1[i])));
      float corr = __expf(mrow[i] - mnew);
      float p0   = __expf(s0[i] - mnew);
      float p1   = __expf(s1[i] - mnew);
      lrow[i] = lrow[i] * corr + rsum16(p0 + p1);
      mrow[i] = mnew;
#pragma unroll
      for (int dt = 0; dt < 8; ++dt) acc[dt][i] *= corr;

      P[(r0 + i) * 32 + cn]      = f2bf(p0);   // per-wave LDS tile; DS ops
      P[(r0 + i) * 32 + 16 + cn] = f2bf(p1);   // are in-order within a wave
    }
    // ---- acc += P(16x32) . V(32xHD) ----------------------------------
    bf16x16 pf = load_frag_a(P, 32);
#pragma unroll
    for (int dt = 0; dt < 8; ++dt) {
      bf16x16 vf = load_frag_b(Vbase + (size_t)(dt * 16) * S + kb, S);
      acc[dt] = wmma_bf16(pf, vf, acc[dt]);
    }
  }

  // ---- normalize + store bf16 attn-out [S][NH*HD] --------------------
#pragma unroll
  for (int i = 0; i < 8; ++i) {
    float inv = 1.0f / lrow[i];
    int row = qbase + r0 + i;
#pragma unroll
    for (int dt = 0; dt < 8; ++dt) {
      Ob[(size_t)row * NQ + h * HD + dt * 16 + cn] = f2bf(acc[dt][i] * inv);
    }
  }
}

// ---------------------------------------------------------------------------
// Host launcher
// ---------------------------------------------------------------------------
extern "C" void kernel_launch(void* const* d_in, const int* in_sizes, int n_in,
                              void* d_out, int out_size, void* d_ws, size_t ws_size,
                              hipStream_t stream) {
  (void)in_sizes; (void)n_in; (void)out_size; (void)ws_size;
  const float* hidden = (const float*)d_in[0];
  const float* cosb   = (const float*)d_in[1];
  const float* sinb   = (const float*)d_in[2];
  // d_in[3] = attention_mask: the reference mask is exactly causal -1e9; we
  // apply it analytically inside flash_attn.
  const float* Wq = (const float*)d_in[4];
  const float* Wk = (const float*)d_in[5];
  const float* Wv = (const float*)d_in[6];
  const float* Wo = (const float*)d_in[7];
  float* out = (float*)d_out;

  char* ws = (char*)d_ws;
  size_t off = 0;
  auto alloc = [&](size_t bytes) -> void* {
    void* p = ws + off;
    off += (bytes + 255) & ~(size_t)255;
    return p;
  };
  unsigned short* Xb   = (unsigned short*)alloc((size_t)S * HID * 2);
  unsigned short* Wqb  = (unsigned short*)alloc((size_t)NQ * HID * 2);
  unsigned short* Wkb  = (unsigned short*)alloc((size_t)NKVD * HID * 2);
  unsigned short* Wvb  = (unsigned short*)alloc((size_t)NKVD * HID * 2);
  unsigned short* Wob  = (unsigned short*)alloc((size_t)HID * NQ * 2);
  unsigned short* Qlin = (unsigned short*)alloc((size_t)S * NQ * 2);
  unsigned short* Klin = (unsigned short*)alloc((size_t)S * NKVD * 2);
  unsigned short* Vlin = (unsigned short*)alloc((size_t)S * NKVD * 2);
  unsigned short* Qh   = (unsigned short*)alloc((size_t)NH * S * HD * 2);
  unsigned short* Kh   = (unsigned short*)alloc((size_t)NKV * S * HD * 2);
  unsigned short* Vt   = (unsigned short*)alloc((size_t)NKV * HD * S * 2);
  unsigned short* Ob   = (unsigned short*)alloc((size_t)S * NQ * 2);

  auto cvt = [&](const float* src, unsigned short* dst, int n) {
    cvt_f32_bf16<<<(n + 255) / 256, 256, 0, stream>>>(src, dst, n);
  };
  cvt(hidden, Xb, S * HID);
  cvt(Wq, Wqb, NQ * HID);
  cvt(Wk, Wkb, NKVD * HID);
  cvt(Wv, Wvb, NKVD * HID);
  cvt(Wo, Wob, HID * NQ);

  // QKV projections (bf16 out)
  gemm_xwt<1><<<dim3(NQ / 128,   S / 64), 256, 0, stream>>>(Xb, Wqb, Qlin, S, NQ,   HID);
  gemm_xwt<1><<<dim3(NKVD / 128, S / 64), 256, 0, stream>>>(Xb, Wkb, Klin, S, NKVD, HID);
  gemm_xwt<1><<<dim3(NKVD / 128, S / 64), 256, 0, stream>>>(Xb, Wvb, Vlin, S, NKVD, HID);

  // RoPE + attention layouts
  rope_layout<<<(S * NQ + 255) / 256, 256, 0, stream>>>(Qlin, cosb, sinb, Qh, NH);
  rope_layout<<<(S * NKVD + 255) / 256, 256, 0, stream>>>(Klin, cosb, sinb, Kh, NKV);
  v_layout<<<(S * NKVD + 255) / 256, 256, 0, stream>>>(Vlin, Vt);

  // Flash attention
  flash_attn<<<dim3(NH, S / 64), 128, 0, stream>>>(Qh, Kh, Vt, Ob);

  // Output projection (fp32 out)
  gemm_xwt<0><<<dim3(HID / 128, S / 64), 256, 0, stream>>>(Ob, Wob, out, S, HID, NQ);
}
